// Depth_prompt_69389491634713
// MI455X (gfx1250) — compile-verified, matrix operand order checked by reading, not run on hardware
//
#include <hip/hip_runtime.h>
#include <math.h>

// ---------------- Types for CDNA5 WMMA ----------------
typedef __attribute__((ext_vector_type(16))) __bf16 v16bf;
typedef __attribute__((ext_vector_type(8)))  float  v8f;

// Problem constants
#define NB    8
#define HH    64
#define WW    64
#define NPIX  4096            // H*W
#define ED    768
#define HID   384
#define LD    24
#define KK    9
#define OW    216             // LD*KK
#define MTOT  32768           // NB*NPIX
#define DEPTHL 4

// float -> bf16, round-to-nearest-even
__device__ __forceinline__ unsigned short f2bf_bits(float f) {
  union { float f; unsigned u; } q; q.f = f;
  unsigned r = q.u + 0x7FFFu + ((q.u >> 16) & 1u);
  return (unsigned short)(r >> 16);
}
// pack two bf16 (k even in low half, k odd in high half) into one dword
__device__ __forceinline__ unsigned pack2bf(float lo, float hi) {
  return (unsigned)f2bf_bits(lo) | ((unsigned)f2bf_bits(hi) << 16);
}

// 16-bit A/B fragment K index for half-element j (0..15) of this lane
// (CDNA5 ISA "16-bit A-Matrix 16x32" layout; B mirrors with N for M).
// Per lane this maps to two contiguous 16-byte LDS runs -> ds_load_b128 x2.
__device__ __forceinline__ int wmma_k_of(int lane, int j) {
  int kbase = (lane < 16) ? 0 : 8;
  int pair  = j >> 1;
  int lo    = j & 1;
  int k0    = (pair < 4) ? (kbase + 2 * pair) : (16 + kbase + 2 * (pair - 4));
  return k0 + lo;
}

// =====================================================================
// Kernel 1: weights[n,o,p] = sigmoid( sum_c depth[n,c,p]*regW[o,c] + regb[o] )
// GEMM M=32768 (pixels), K=768, N=216 (pad 224). Block = 32-pixel M-tile,
// 14 waves x (16-wide N sub-tile, two 16-row M sub-tiles). bf16 WMMA.
// =====================================================================
__global__ void k_reg_gemm(const float* __restrict__ depth,
                           const float* __restrict__ regW,
                           const float* __restrict__ regb,
                           float* __restrict__ wout) {
  __shared__ unsigned sA32[32 * 16];     // 32 rows x 32 bf16  [m][k]
  __shared__ unsigned sB32[224 * 16];    // 224 rows x 32 bf16 [o][k]
  const __bf16* sA = (const __bf16*)sA32;
  const __bf16* sB = (const __bf16*)sB32;
  const int tid  = threadIdx.x;          // 448 threads = 14 waves
  const int lane = tid & 31;
  const int wv   = tid >> 5;
  const int p0   = blockIdx.x * 32;
  const int n    = p0 >> 12;
  const int pix0 = p0 & (NPIX - 1);

  v8f acc0 = {}, acc1 = {};
  for (int kc = 0; kc < ED / 32; ++kc) {
    const int k0 = kc * 32;
    // stage A: 32 pixels x 16 k-pairs; coalesced along pixels
    for (int idx = tid; idx < 32 * 16; idx += 448) {
      int m = idx & 31, kp = idx >> 5;
      size_t base = ((size_t)(n * ED + k0 + 2 * kp)) * NPIX + pix0 + m;
      sA32[m * 16 + kp] = pack2bf(depth[base], depth[base + NPIX]);
    }
    // stage B: 224 outputs x 16 k-pairs; contiguous float2 along k
    for (int idx = tid; idx < 224 * 16; idx += 448) {
      int o = idx >> 4, kp = idx & 15;
      float lo = 0.0f, hi = 0.0f;
      if (o < OW) {
        const float2 w2 = *(const float2*)&regW[(size_t)o * ED + k0 + 2 * kp];
        lo = w2.x; hi = w2.y;
      }
      sB32[o * 16 + kp] = pack2bf(lo, hi);
    }
    __syncthreads();

    v16bf a0, a1, b;
    const __bf16* rA0 = &sA[(lane & 15) * 32];
    const __bf16* rA1 = &sA[(16 + (lane & 15)) * 32];
    const __bf16* rB  = &sB[(wv * 16 + (lane & 15)) * 32];
#pragma unroll
    for (int j = 0; j < 16; ++j) {
      int k = wmma_k_of(lane, j);
      a0[j] = rA0[k];
      a1[j] = rA1[k];
      b[j]  = rB[k];
    }
    acc0 = __builtin_amdgcn_wmma_f32_16x16x32_bf16(
        false, a0, false, b, (short)0, acc0, false, false);
    acc1 = __builtin_amdgcn_wmma_f32_16x16x32_bf16(
        false, a1, false, b, (short)0, acc1, false, false);
    __syncthreads();
  }

  const int o    = wv * 16 + (lane & 15);
  const int mofs = (lane < 16) ? 0 : 8;
  if (o < OW) {
    const float bias = regb[o];
    const size_t ob = ((size_t)n * OW + o) * NPIX + pix0;
#pragma unroll
    for (int vj = 0; vj < 8; ++vj) {
      int m = vj + mofs;
      float x0 = acc0[vj] + bias;
      float x1 = acc1[vj] + bias;
      wout[ob + m]      = 1.0f / (1.0f + expf(-x0));
      wout[ob + 16 + m] = 1.0f / (1.0f + expf(-x1));
    }
  }
}

// =====================================================================
// Kernel 2: in-place per-pixel stencil normalization over the 9 taps
// =====================================================================
__global__ void k_normalize(float* __restrict__ w) {
  int t = blockIdx.x * blockDim.x + threadIdx.x;  // NB*LD*NPIX = 786432
  int pix = t & (NPIX - 1);
  int r = t >> 12;
  int ld = r % LD, n = r / LD;
  size_t base = ((size_t)n * OW + ld * KK) * NPIX + pix;
  float vals[KK];
  float s = 0.0f;
#pragma unroll
  for (int k = 0; k < KK; ++k) { vals[k] = w[base + (size_t)k * NPIX]; s += vals[k]; }
  float inv = 1.0f / (s + 1e-5f);
#pragma unroll
  for (int k = 0; k < KK; ++k) w[base + (size_t)k * NPIX] = vals[k] * inv;
}

// =====================================================================
// Kernel 3: generic 3x3 conv, stride 1, pad 1, NCHW/OIHW, optional ReLU
// =====================================================================
__global__ void k_conv3(const float* __restrict__ in,
                        const float* __restrict__ Wk,
                        const float* __restrict__ b,
                        float* __restrict__ out,
                        int Cin, int Cout, int relu) {
  int t = blockIdx.x * blockDim.x + threadIdx.x;
  int total = NB * Cout * NPIX;
  if (t >= total) return;
  int pix = t & (NPIX - 1);
  int r = t >> 12;
  int co = r % Cout, n = r / Cout;
  int hh = pix >> 6, wc = pix & 63;
  float acc = b[co];
  for (int ci = 0; ci < Cin; ++ci) {
    const float* ip = in + ((size_t)n * Cin + ci) * NPIX;
    const float* wp = Wk + ((size_t)(co * Cin + ci)) * 9;
#pragma unroll
    for (int di = -1; di <= 1; ++di) {
      int hy = hh + di;
      if ((unsigned)hy >= (unsigned)HH) continue;
#pragma unroll
      for (int dj = -1; dj <= 1; ++dj) {
        int wx = wc + dj;
        if ((unsigned)wx >= (unsigned)WW) continue;
        acc += ip[hy * WW + wx] * wp[(di + 1) * 3 + (dj + 1)];
      }
    }
  }
  if (relu) acc = fmaxf(acc, 0.0f);
  out[t] = acc;
}

// =====================================================================
// Kernel 4: one message-passing step: out = sum_k unfold3(in)*normw
// =====================================================================
__global__ void k_msgpass(const float* __restrict__ in,
                          const float* __restrict__ nw,
                          float* __restrict__ out) {
  int t = blockIdx.x * blockDim.x + threadIdx.x;  // 786432
  int pix = t & (NPIX - 1);
  int r = t >> 12;
  int ld = r % LD, n = r / LD;
  int hh = pix >> 6, wc = pix & 63;
  const float* ip = in + ((size_t)n * LD + ld) * NPIX;
  size_t wb = ((size_t)n * OW + ld * KK) * NPIX + pix;
  float acc = 0.0f;
  int k = 0;
#pragma unroll
  for (int di = -1; di <= 1; ++di)
#pragma unroll
    for (int dj = -1; dj <= 1; ++dj, ++k) {
      int hy = hh + di, wx = wc + dj;
      if ((unsigned)hy < (unsigned)HH && (unsigned)wx < (unsigned)WW)
        acc += ip[hy * WW + wx] * nw[wb + (size_t)k * NPIX];
    }
  out[t] = acc;
}

// =====================================================================
// Kernel 5: rank-1 collapse of the layer MLPs:
//   u[i][j'] = lmlp_W[i][j'] . da_W ;  v[i][j'] = lmlp_W[i][j'] . da_b + lmlp_b[i][j']
// =====================================================================
__global__ void k_layer_vec(const float* __restrict__ lW,
                            const float* __restrict__ lb,
                            const float* __restrict__ daW,
                            const float* __restrict__ dab,
                            float* __restrict__ u, float* __restrict__ v) {
  int t = blockIdx.x * blockDim.x + threadIdx.x;
  if (t >= DEPTHL * HID) return;
  int i = t / HID, jp = t % HID;
  const float* wr = lW + ((size_t)i * HID + jp) * HID;
  float su = 0.0f, sv = 0.0f;
  for (int j = 0; j < HID; ++j) {
    float w = wr[j];
    su += w * daW[j];
    sv += w * dab[j];
  }
  u[t] = su;
  v[t] = sv + lb[i * HID + jp];
}

// =====================================================================
// Kernel 6: prompts GEMM (dominant FLOPs + the 402MB output stream):
//   out[i,p,e] = sum_j gelu(s[p]*u_i[j] + v_i[j]) * sm_W[e,j] + sm_b[e]
// M=32768, K=384, N=768, 4 layers. Block = 32-pixel M-tile x 256-wide N
// panel; 16 waves x (16-wide N sub-tile, two 16-row M sub-tiles).
// =====================================================================
__global__ void k_prompts(const float* __restrict__ s,
                          const float* __restrict__ u,
                          const float* __restrict__ v,
                          const float* __restrict__ smW,
                          const float* __restrict__ smb,
                          float* __restrict__ out) {
  __shared__ float    s_s[32];
  __shared__ float    s_u[HID];
  __shared__ float    s_v[HID];
  __shared__ unsigned sA32[32 * 16];     // 32 rows x 32 bf16  [m][k]
  __shared__ unsigned sB32[256 * 16];    // 256 rows x 32 bf16 [e_local][k]
  const __bf16* sA = (const __bf16*)sA32;
  const __bf16* sB = (const __bf16*)sB32;
  const int tid    = threadIdx.x;        // 512 threads = 16 waves
  const int lane   = tid & 31;
  const int wv     = tid >> 5;
  const int p0     = blockIdx.x * 32;
  const int e0base = blockIdx.y * 256;
  const int layer  = blockIdx.z;

  if (tid < 32) s_s[tid] = s[p0 + tid];
  for (int idx = tid; idx < HID; idx += 512) {
    s_u[idx] = u[layer * HID + idx];
    s_v[idx] = v[layer * HID + idx];
  }
  __syncthreads();

  v8f acc0 = {}, acc1 = {};
  for (int kc = 0; kc < HID / 32; ++kc) {
    const int k0 = kc * 32;
    // prefetch next B K-slab into GL2 while we work on this one
    if (kc + 1 < HID / 32)
      __builtin_prefetch(&smW[(size_t)(e0base + (tid >> 1)) * HID + k0 + 32], 0, 1);
    // stage A: gelu(s[m]*u[k]+v[k]) -> packed bf16 pairs (exact erf gelu)
    {
      int m = tid & 31, kp = tid >> 5;   // 512 threads = 32 x 16 pairs
      int ka = k0 + 2 * kp;
      float x0 = s_s[m] * s_u[ka]     + s_v[ka];
      float x1 = s_s[m] * s_u[ka + 1] + s_v[ka + 1];
      float g0 = 0.5f * x0 * (1.0f + erff(x0 * 0.70710678f));
      float g1 = 0.5f * x1 * (1.0f + erff(x1 * 0.70710678f));
      sA32[m * 16 + kp] = pack2bf(g0, g1);
    }
    // stage B: sm_W[e, k] contiguous float2 along k
    for (int idx = tid; idx < 256 * 16; idx += 512) {
      int el = idx >> 4, kp = idx & 15;
      const float2 w2 =
          *(const float2*)&smW[(size_t)(e0base + el) * HID + k0 + 2 * kp];
      sB32[el * 16 + kp] = pack2bf(w2.x, w2.y);
    }
    __syncthreads();

    v16bf a0, a1, b;
    const __bf16* rA0 = &sA[(lane & 15) * 32];
    const __bf16* rA1 = &sA[(16 + (lane & 15)) * 32];
    const __bf16* rB  = &sB[(wv * 16 + (lane & 15)) * 32];
#pragma unroll
    for (int j = 0; j < 16; ++j) {
      int k = wmma_k_of(lane, j);
      a0[j] = rA0[k];
      a1[j] = rA1[k];
      b[j]  = rB[k];
    }
    acc0 = __builtin_amdgcn_wmma_f32_16x16x32_bf16(
        false, a0, false, b, (short)0, acc0, false, false);
    acc1 = __builtin_amdgcn_wmma_f32_16x16x32_bf16(
        false, a1, false, b, (short)0, acc1, false, false);
    __syncthreads();
  }

  const int e    = e0base + wv * 16 + (lane & 15);
  const int mofs = (lane < 16) ? 0 : 8;
  const float bias = smb[e];
  const size_t obase = (size_t)layer * MTOT * ED + (size_t)p0 * ED + e;
#pragma unroll
  for (int vj = 0; vj < 8; ++vj) {
    int m = vj + mofs;
    out[obase + (size_t)m * ED]        = acc0[vj] + bias;  // rows p0+m
    out[obase + (size_t)(16 + m) * ED] = acc1[vj] + bias;  // rows p0+16+m
  }
}

// =====================================================================
extern "C" void kernel_launch(void* const* d_in, const int* in_sizes, int n_in,
                              void* d_out, int out_size, void* d_ws, size_t ws_size,
                              hipStream_t stream) {
  (void)in_sizes; (void)n_in; (void)out_size; (void)ws_size;
  const float* depth  = (const float*)d_in[0];
  const float* cues   = (const float*)d_in[1];
  // d_in[2], d_in[3] (eg_W/eg_b) are dead code in the reference
  const float* sm_W   = (const float*)d_in[4];
  const float* sm_b   = (const float*)d_in[5];
  const float* da_W   = (const float*)d_in[6];
  const float* da_b   = (const float*)d_in[7];
  const float* lmlp_W = (const float*)d_in[8];
  const float* lmlp_b = (const float*)d_in[9];
  const float* reg_W  = (const float*)d_in[10];
  const float* reg_b  = (const float*)d_in[11];
  const float* eW0 = (const float*)d_in[12]; const float* eb0 = (const float*)d_in[13];
  const float* eW1 = (const float*)d_in[14]; const float* eb1 = (const float*)d_in[15];
  const float* eW2 = (const float*)d_in[16]; const float* eb2 = (const float*)d_in[17];
  const float* dW0 = (const float*)d_in[18]; const float* db0 = (const float*)d_in[19];
  const float* dW1 = (const float*)d_in[20]; const float* db1 = (const float*)d_in[21];
  const float* dW2 = (const float*)d_in[22]; const float* db2 = (const float*)d_in[23];
  float* out = (float*)d_out;

  // workspace layout (floats)
  float* ws    = (float*)d_ws;
  float* w_buf = ws;                                    // NB*OW*NPIX   = 7,077,888
  float* B1    = w_buf + (size_t)NB * OW * NPIX;        // NB*LD*NPIX   =   786,432
  float* B2    = B1 + (size_t)NB * LD * NPIX;           //   786,432
  float* sbuf  = B2 + (size_t)NB * LD * NPIX;           //    32,768
  float* ubuf  = sbuf + MTOT;                           //     1,536
  float* vbuf  = ubuf + DEPTHL * HID;                   //     1,536

  // 1) propagation-weight GEMM + sigmoid  (WMMA)
  k_reg_gemm<<<dim3(MTOT / 32), dim3(448), 0, stream>>>(depth, reg_W, reg_b, w_buf);
  // 2) per-pixel stencil normalization (in place)
  k_normalize<<<dim3(3072), dim3(256), 0, stream>>>(w_buf);
  // 3) encoder
  k_conv3<<<dim3(3072), dim3(256), 0, stream>>>(cues, eW0, eb0, B1, 1,  LD, 1);
  k_conv3<<<dim3(3072), dim3(256), 0, stream>>>(B1,   eW1, eb1, B2, LD, LD, 1);
  k_conv3<<<dim3(3072), dim3(256), 0, stream>>>(B2,   eW2, eb2, B1, LD, LD, 0);
  // 4) 7-step stencil diffusion (ping-pong B1<->B2, ends in B2)
  float* pin = B1; float* pout = B2;
  for (int step = 0; step < 7; ++step) {
    k_msgpass<<<dim3(3072), dim3(256), 0, stream>>>(pin, w_buf, pout);
    float* t = pin; pin = pout; pout = t;
  }
  // 5) decoder -> shape activation s (NB*NPIX)
  k_conv3<<<dim3(3072), dim3(256), 0, stream>>>(pin,  dW0, db0, pout, LD, LD, 1);
  k_conv3<<<dim3(3072), dim3(256), 0, stream>>>(pout, dW1, db1, pin,  LD, LD, 1);
  k_conv3<<<dim3(128),  dim3(256), 0, stream>>>(pin,  dW2, db2, sbuf, LD, 1,  0);
  // 6) rank-1 collapse of layer MLPs
  k_layer_vec<<<dim3(6), dim3(256), 0, stream>>>(lmlp_W, lmlp_b, da_W, da_b, ubuf, vbuf);
  // 7) prompt GEMM (WMMA): out[layer, p, e]
  k_prompts<<<dim3(MTOT / 32, ED / 256, DEPTHL), dim3(512), 0, stream>>>(
      sbuf, ubuf, vbuf, sm_W, sm_b, out);
}
